// AttentionBlock_1408749273213
// MI455X (gfx1250) — matrix operand order repaired and checked by reading, not compiled, not run on hardware
//
#include <hip/hip_runtime.h>
#include <hip/hip_bf16.h>
#include <stdint.h>

// ---------------------------------------------------------------------------
// AttentionBlock for MI455X (gfx1250, wave32, WMMA 16x16x32 f16 -> f32 acc)
// B=16, C=512, H=W=32 -> N=1024 tokens, 4 heads, d_k=128
// ---------------------------------------------------------------------------

#define BATCH   16
#define CDIM    512
#define NTOK    1024
#define NH      4
#define DK      128
#define C3      (3 * CDIM)          // 1536
#define H3DK    (3 * DK)            // 384, per-head stride inside qkv row

#define KSTRIDE 136                 // K tile LDS row stride (halves): 272B = 17*16
#define VSTRIDE 34                  // V^T / P LDS row stride (halves)

typedef __attribute__((ext_vector_type(16))) _Float16 v16h;
typedef __attribute__((ext_vector_type(8)))  float    v8f;
typedef int v4i_b128 __attribute__((vector_size(16)));   // async b128 payload type

union FragH { v16h h; unsigned u[8]; };

#if __has_builtin(__builtin_amdgcn_global_load_async_to_lds_b128) && \
    __has_builtin(__builtin_amdgcn_s_wait_asynccnt)
#define USE_ASYNC_LDS 1
#else
#define USE_ASYNC_LDS 0
#endif

#if USE_ASYNC_LDS
__device__ __forceinline__ void async_b128(const void* g, void* lds) {
    __builtin_amdgcn_global_load_async_to_lds_b128(
        (__attribute__((address_space(1))) v4i_b128*)(v4i_b128*)(void*)g,
        (__attribute__((address_space(3))) v4i_b128*)(v4i_b128*)lds,
        0, 0);
}
#endif

// A-matrix (16x32 f16) K offset for VGPR v, lane-half: K pairs {0..7,16..23} / {8..15,24..31}
__device__ __forceinline__ int kbA(int v, int half) {
    return ((v & 3) << 1) + ((v >> 2) << 4) + (half << 3);
}
// B-matrix (32x16 f16) K offset, same as A: K pairs {0..7,16..23} / {8..15,24..31}
__device__ __forceinline__ int kbB(int v, int half) {
    return ((v & 3) << 1) + ((v >> 2) << 4) + (half << 3);
}

__device__ __forceinline__ v8f wmma16(const FragH& a, const FragH& b, v8f c) {
    return __builtin_amdgcn_wmma_f32_16x16x32_f16(
        /*neg_a=*/false, a.h, /*neg_b=*/false, b.h,
        /*c_mod=*/(short)0, c, /*reuse_a=*/false, /*reuse_b=*/false);
}

__device__ __forceinline__ v8f zero8() {
    v8f z = {0.f, 0.f, 0.f, 0.f, 0.f, 0.f, 0.f, 0.f};
    return z;
}

// ---------------------------------------------------------------------------
// 1) x (B,C,N) f32  ->  xf16 (B,N,C) f16   (coalesced reads along N)
// ---------------------------------------------------------------------------
__global__ void cvt_x_kernel(const float* __restrict__ x, _Float16* __restrict__ xf16) {
    size_t tid = (size_t)blockIdx.x * blockDim.x + threadIdx.x;
    if (tid >= (size_t)BATCH * CDIM * NTOK) return;
    int n = (int)(tid & (NTOK - 1));
    int c = (int)((tid >> 10) & (CDIM - 1));
    int b = (int)(tid >> 19);
    xf16[((size_t)(b * NTOK + n)) * CDIM + c] = (_Float16)x[tid];
}

// ---------------------------------------------------------------------------
// 2) weights -> transposed f16: wpT[nOut][k], woT[nOut][k]
// ---------------------------------------------------------------------------
__global__ void cvt_w_kernel(const float* __restrict__ w_proj, const float* __restrict__ w_out,
                             _Float16* __restrict__ wpT, _Float16* __restrict__ woT) {
    size_t tid = (size_t)blockIdx.x * blockDim.x + threadIdx.x;
    const size_t NP = (size_t)CDIM * C3;
    if (tid < NP) {
        int n = (int)(tid % C3);
        int k = (int)(tid / C3);
        wpT[(size_t)n * CDIM + k] = (_Float16)w_proj[tid];
    } else if (tid < NP + (size_t)CDIM * CDIM) {
        size_t t2 = tid - NP;
        int n = (int)(t2 % CDIM);
        int k = (int)(t2 / CDIM);
        woT[(size_t)n * CDIM + k] = (_Float16)w_out[t2];
    }
}

// ---------------------------------------------------------------------------
// 3) QKV GEMM: one wave -> 64(M) x 64(N) tile, K=512, 16 WMMAs per k-step
// ---------------------------------------------------------------------------
__global__ __launch_bounds__(256) void qkv_gemm_kernel(
    const _Float16* __restrict__ xf16, const _Float16* __restrict__ wpT,
    const float* __restrict__ b_proj, _Float16* __restrict__ qkv16) {

    const int MT = NTOK / 64;   // 16
    const int NG = C3 / 64;     // 24
    int wid  = blockIdx.x * (blockDim.x >> 5) + (threadIdx.x >> 5);
    int lane = threadIdx.x & 31;
    int mt = wid % MT;
    int t  = wid / MT;
    int ng = t % NG;
    int b  = t / NG;
    int mbase = mt * 64, nbase = ng * 64;
    int r = lane & 15, half = lane >> 4;

    v8f acc[4][4];
#pragma unroll
    for (int ms = 0; ms < 4; ++ms)
#pragma unroll
        for (int ns = 0; ns < 4; ++ns) acc[ms][ns] = zero8();

    for (int k0 = 0; k0 < CDIM; k0 += 32) {
        FragH a[4];
#pragma unroll
        for (int ms = 0; ms < 4; ++ms) {
            const _Float16* arow =
                xf16 + ((size_t)(b * NTOK + mbase + ms * 16 + r)) * CDIM + k0;
#pragma unroll
            for (int v = 0; v < 8; ++v)
                a[ms].u[v] = *(const unsigned*)(arow + kbA(v, half));
        }
        FragH bf[4];
#pragma unroll
        for (int ns = 0; ns < 4; ++ns) {
            const _Float16* brow =
                wpT + ((size_t)(nbase + ns * 16 + r)) * CDIM + k0;
#pragma unroll
            for (int v = 0; v < 8; ++v)
                bf[ns].u[v] = *(const unsigned*)(brow + kbB(v, half));
        }
#pragma unroll
        for (int ms = 0; ms < 4; ++ms)
#pragma unroll
            for (int ns = 0; ns < 4; ++ns)
                acc[ms][ns] = wmma16(a[ms], bf[ns], acc[ms][ns]);
    }
#pragma unroll
    for (int ns = 0; ns < 4; ++ns) {
        int col = nbase + ns * 16 + r;
        float bias = b_proj[col];
#pragma unroll
        for (int ms = 0; ms < 4; ++ms)
#pragma unroll
            for (int v = 0; v < 8; ++v) {
                int m = mbase + ms * 16 + v + (half << 3);
                qkv16[((size_t)(b * NTOK + m)) * C3 + col] = (_Float16)(acc[ms][ns][v] + bias);
            }
    }
}

// ---------------------------------------------------------------------------
// 4) Flash attention: per (b,h), 64 queries per workgroup (4 waves x 16 rows)
//    K tile async-staged to LDS (ASYNCcnt path), V manually transposed in LDS
// ---------------------------------------------------------------------------
__global__ __launch_bounds__(128) void attn_kernel(
    const _Float16* __restrict__ qkv16, _Float16* __restrict__ res16) {

    __shared__ unsigned short sK[32 * KSTRIDE];      // K tile [j][d], 272B rows (16B aligned)
    __shared__ unsigned short sV[128 * VSTRIDE];     // V tile transposed [d][j]
    __shared__ unsigned short sP[4 * 16 * VSTRIDE];  // per-wave P scratch [m][j']

    int ig = blockIdx.x, h = blockIdx.y, b = blockIdx.z;
    int wave = threadIdx.x >> 5, lane = threadIdx.x & 31;
    int r = lane & 15, half = lane >> 4;
    int ibase = ig * 64 + wave * 16;
    const float scale = 0.088388347648318447f;       // 128^-0.5

    const size_t hbase = ((size_t)(b * NTOK)) * C3 + h * H3DK;

    // Q fragments for all 4 K-chunks of d_k=128 (A operand, rows = queries)
    FragH aq[4];
    const _Float16* qrow = qkv16 + hbase + (size_t)(ibase + r) * C3;
#pragma unroll
    for (int ch = 0; ch < 4; ++ch)
#pragma unroll
        for (int v = 0; v < 8; ++v)
            aq[ch].u[v] = *(const unsigned*)(qrow + ch * 32 + kbA(v, half));

    float rmax[8], rsum[8];
    v8f acc[8];
#pragma unroll
    for (int v = 0; v < 8; ++v) { rmax[v] = -1e30f; rsum[v] = 0.f; }
#pragma unroll
    for (int d = 0; d < 8; ++d) acc[d] = zero8();

    unsigned short* myP = sP + wave * 16 * VSTRIDE;

    for (int jb = 0; jb < NTOK; jb += 32) {
        __syncthreads();   // previous tile fully consumed by all waves

#if USE_ASYNC_LDS
        // K tile: async copy global -> LDS (16B chunks, per-lane LDS scatter keeps padding)
        for (int u = threadIdx.x; u < 32 * 16; u += 128) {
            int j  = u >> 4;          // row 0..31
            int ch = u & 15;          // 16B chunk 0..15
            const _Float16* g = qkv16 + hbase + (size_t)(jb + j) * C3 + DK + ch * 8;
            async_b128(g, (char*)sK + j * (KSTRIDE * 2) + ch * 16);
        }
        // V tile: load + transpose through VGPRs
        for (int idx = threadIdx.x; idx < 32 * 64; idx += 128) {
            int j  = idx >> 6;
            int du = (idx & 63) << 1;
            const _Float16* row = qkv16 + hbase + (size_t)(jb + j) * C3;
            unsigned vv = *(const unsigned*)(row + 2 * DK + du);
            sV[(du + 0) * VSTRIDE + j] = (unsigned short)(vv & 0xffffu);
            sV[(du + 1) * VSTRIDE + j] = (unsigned short)(vv >> 16);
        }
        __builtin_amdgcn_s_wait_asynccnt(0);
#else
        for (int idx = threadIdx.x; idx < 32 * 64; idx += 128) {
            int j  = idx >> 6;
            int du = (idx & 63) << 1;
            const _Float16* row = qkv16 + hbase + (size_t)(jb + j) * C3;
            unsigned kk = *(const unsigned*)(row + DK + du);
            *(unsigned*)&sK[j * KSTRIDE + du] = kk;
            unsigned vv = *(const unsigned*)(row + 2 * DK + du);
            sV[(du + 0) * VSTRIDE + j] = (unsigned short)(vv & 0xffffu);
            sV[(du + 1) * VSTRIDE + j] = (unsigned short)(vv >> 16);
        }
#endif
        __syncthreads();

        // prefetch next j-tile's K/V rows while computing
        if (jb + 32 < NTOK) {
            int j  = threadIdx.x >> 2;                 // 0..31
            int ch = (threadIdx.x & 3) * 64;           // 4 chunks over 256 halves (K+V)
            __builtin_prefetch(qkv16 + hbase + (size_t)(jb + 32 + j) * C3 + DK + ch, 0, 3);
        }

        // S = Q @ K^T  (two 16x16 column tiles, accumulate over 4 d-chunks)
        v8f s0 = zero8(), s1 = zero8();
#pragma unroll
        for (int ch = 0; ch < 4; ++ch) {
            FragH bk;
#pragma unroll
            for (int v = 0; v < 8; ++v)
                bk.u[v] = *(const unsigned*)&sK[r * KSTRIDE + ch * 32 + kbB(v, half)];
            s0 = wmma16(aq[ch], bk, s0);
        }
#pragma unroll
        for (int ch = 0; ch < 4; ++ch) {
            FragH bk;
#pragma unroll
            for (int v = 0; v < 8; ++v)
                bk.u[v] = *(const unsigned*)&sK[(16 + r) * KSTRIDE + ch * 32 + kbB(v, half)];
            s1 = wmma16(aq[ch], bk, s1);
        }

        // online softmax per row (row m = v + 8*half; 16 lanes hold the columns)
#pragma unroll
        for (int v = 0; v < 8; ++v) {
            float a0 = s0[v] * scale;
            float a1 = s1[v] * scale;
            float mx = fmaxf(a0, a1);
            mx = fmaxf(mx, __shfl_xor(mx, 1));
            mx = fmaxf(mx, __shfl_xor(mx, 2));
            mx = fmaxf(mx, __shfl_xor(mx, 4));
            mx = fmaxf(mx, __shfl_xor(mx, 8));
            float nmax  = fmaxf(rmax[v], mx);
            float alpha = __expf(rmax[v] - nmax);
            float p0 = __expf(a0 - nmax);
            float p1 = __expf(a1 - nmax);
            float ps = p0 + p1;
            ps += __shfl_xor(ps, 1);
            ps += __shfl_xor(ps, 2);
            ps += __shfl_xor(ps, 4);
            ps += __shfl_xor(ps, 8);
            rsum[v] = rsum[v] * alpha + ps;
            rmax[v] = nmax;
            int m = v + (half << 3);
            *(_Float16*)&myP[m * VSTRIDE + r]      = (_Float16)p0;
            *(_Float16*)&myP[m * VSTRIDE + 16 + r] = (_Float16)p1;
#pragma unroll
            for (int d = 0; d < 8; ++d) acc[d][v] *= alpha;
        }

        // O += P @ V   (A = P via LDS reshape; B = transposed V tile)
        FragH ap;
#pragma unroll
        for (int v = 0; v < 8; ++v)
            ap.u[v] = *(const unsigned*)&myP[r * VSTRIDE + kbA(v, half)];
#pragma unroll
        for (int d = 0; d < 8; ++d) {
            FragH bv;
#pragma unroll
            for (int v = 0; v < 8; ++v)
                bv.u[v] = *(const unsigned*)&sV[(d * 16 + r) * VSTRIDE + kbB(v, half)];
            acc[d] = wmma16(ap, bv, acc[d]);
        }
    }

    // normalize + store res16 (B,N,C) with C = head-major channels
#pragma unroll
    for (int v = 0; v < 8; ++v) {
        float inv = 1.f / rsum[v];
        int m = v + (half << 3);
        size_t rowo = ((size_t)(b * NTOK + ibase + m)) * CDIM + h * DK;
#pragma unroll
        for (int d = 0; d < 8; ++d)
            res16[rowo + d * 16 + r] = (_Float16)(acc[d][v] * inv);
    }
}

// ---------------------------------------------------------------------------
// 5) Out GEMM (64x64 wave tiles) + bias + residual + (B,N,C)->(B,C,N) store
// ---------------------------------------------------------------------------
__global__ __launch_bounds__(256) void out_gemm_kernel(
    const _Float16* __restrict__ res16, const _Float16* __restrict__ woT,
    const float* __restrict__ b_out, const float* __restrict__ x,
    float* __restrict__ out) {

    const int MT = NTOK / 64;   // 16
    const int NG = CDIM / 64;   // 8
    int wid  = blockIdx.x * (blockDim.x >> 5) + (threadIdx.x >> 5);
    int lane = threadIdx.x & 31;
    int mt = wid % MT;
    int t  = wid / MT;
    int ng = t % NG;
    int b  = t / NG;
    int mbase = mt * 64, nbase = ng * 64;
    int r = lane & 15, half = lane >> 4;

    v8f acc[4][4];
#pragma unroll
    for (int ms = 0; ms < 4; ++ms)
#pragma unroll
        for (int ns = 0; ns < 4; ++ns) acc[ms][ns] = zero8();

    for (int k0 = 0; k0 < CDIM; k0 += 32) {
        FragH a[4];
#pragma unroll
        for (int ms = 0; ms < 4; ++ms) {
            const _Float16* arow =
                res16 + ((size_t)(b * NTOK + mbase + ms * 16 + r)) * CDIM + k0;
#pragma unroll
            for (int v = 0; v < 8; ++v)
                a[ms].u[v] = *(const unsigned*)(arow + kbA(v, half));
        }
        FragH bf[4];
#pragma unroll
        for (int ns = 0; ns < 4; ++ns) {
            const _Float16* brow =
                woT + ((size_t)(nbase + ns * 16 + r)) * CDIM + k0;
#pragma unroll
            for (int v = 0; v < 8; ++v)
                bf[ns].u[v] = *(const unsigned*)(brow + kbB(v, half));
        }
#pragma unroll
        for (int ms = 0; ms < 4; ++ms)
#pragma unroll
            for (int ns = 0; ns < 4; ++ns)
                acc[ms][ns] = wmma16(a[ms], bf[ns], acc[ms][ns]);
    }
#pragma unroll
    for (int ns = 0; ns < 4; ++ns) {
        int col = nbase + ns * 16 + r;          // output channel c
        float bias = b_out[col];
#pragma unroll
        for (int ms = 0; ms < 4; ++ms)
#pragma unroll
            for (int v = 0; v < 8; ++v) {
                int tok = mbase + ms * 16 + v + (half << 3);
                size_t oidx = ((size_t)(b * CDIM + col)) * NTOK + tok;
                out[oidx] = acc[ms][ns][v] + bias + x[oidx];
            }
    }
}

// ---------------------------------------------------------------------------
extern "C" void kernel_launch(void* const* d_in, const int* in_sizes, int n_in,
                              void* d_out, int out_size, void* d_ws, size_t ws_size,
                              hipStream_t stream) {
    const float* x      = (const float*)d_in[0];   // (16,512,32,32)
    const float* w_proj = (const float*)d_in[1];   // (512,1536)
    const float* b_proj = (const float*)d_in[2];   // (1536,)
    const float* w_out  = (const float*)d_in[3];   // (512,512)
    const float* b_out  = (const float*)d_in[4];   // (512,)
    float* out = (float*)d_out;

    char* ws = (char*)d_ws;
    const size_t SZ_XF   = (size_t)BATCH * NTOK * CDIM * 2;   // 16 MB
    const size_t SZ_QKV  = (size_t)BATCH * NTOK * C3 * 2;     // 48 MB
    const size_t SZ_WPT  = (size_t)CDIM * C3 * 2;             // 1.5 MB
    _Float16* xf16  = (_Float16*)(ws);                         // also reused as res16
    _Float16* qkv16 = (_Float16*)(ws + SZ_XF);
    _Float16* wpT   = (_Float16*)(ws + SZ_XF + SZ_QKV);
    _Float16* woT   = (_Float16*)(ws + SZ_XF + SZ_QKV + SZ_WPT);
    _Float16* res16 = xf16;   // xf16 dead after qkv_gemm; reuse for attention output

    // 1) convert x (transpose to token-major f16)
    {
        size_t total = (size_t)BATCH * CDIM * NTOK;
        cvt_x_kernel<<<(unsigned)((total + 255) / 256), 256, 0, stream>>>(x, xf16);
    }
    // 2) convert + transpose weights
    {
        size_t total = (size_t)CDIM * C3 + (size_t)CDIM * CDIM;
        cvt_w_kernel<<<(unsigned)((total + 255) / 256), 256, 0, stream>>>(w_proj, w_out, wpT, woT);
    }
    // 3) QKV projection: 16*24*16 = 6144 waves, 8 waves/block
    {
        int waves = (NTOK / 64) * (C3 / 64) * BATCH;
        qkv_gemm_kernel<<<waves / 8, 256, 0, stream>>>(xf16, wpT, b_proj, qkv16);
    }
    // 4) flash attention
    {
        dim3 grid(NTOK / 64, NH, BATCH);               // (16,4,16)
        attn_kernel<<<grid, 128, 0, stream>>>(qkv16, res16);
    }
    // 5) output projection + bias + residual + transpose: 2048 waves
    {
        int waves = (NTOK / 64) * (CDIM / 64) * BATCH;
        out_gemm_kernel<<<waves / 8, 256, 0, stream>>>(res16, woT, b_out, x, out);
    }
}
